// LSTMRegressor_2130303778835
// MI455X (gfx1250) — matrix operand order repaired and checked by reading, not compiled, not run on hardware
//
#include <hip/hip_runtime.h>
#include <hip/hip_bf16.h>

// Problem constants (match reference)
#define Bz 64
#define Tz 512
#define Iz 128
#define Hz 1024
#define Oz 128
#define Gz (4 * Hz)
#define NWG 32
#define NTHREADS 256
#define BH (Bz * Hz) // 65536 elements

// LDS layout: 128KB h-staging + 8KB split-K reduction buffer
#define SH_H_BYTES (BH * 2)
#define SMEM_BYTES (SH_H_BYTES + 8 * 32 * 8 * 4)

typedef __bf16 bf16_t;
typedef bf16_t v16bf __attribute__((ext_vector_type(16)));
typedef float v8f __attribute__((ext_vector_type(8)));
typedef unsigned int v4u __attribute__((ext_vector_type(4)));
typedef int v8i __attribute__((ext_vector_type(8)));
typedef int v4i __attribute__((ext_vector_type(4)));

#if __has_builtin(__builtin_amdgcn_tensor_load_to_lds) && \
    __has_builtin(__builtin_amdgcn_s_wait_tensorcnt)
#define USE_TDM 1
#endif

union Frag {
  v4u q[2];
  v16bf bf;
};

// A-matrix (16x32 bf16, M x K) per-lane fragment from row-major storage
// (global or LDS; addrspace is inferred from the pointer's derivation).
// ISA layout: lane (m = lane&15, hi = lane>>4):
//   halves 0-7  -> k = kbase + 8*hi + 0..7
//   halves 8-15 -> k = kbase + 8*hi + 16..23
__device__ __forceinline__ v16bf load_A_frag(const bf16_t* base, int row,
                                             int rowStride, int kbase,
                                             int hi) {
  const bf16_t* p = base + (size_t)row * rowStride + kbase + 8 * hi;
  Frag f;
  f.q[0] = *(const v4u*)p;
  f.q[1] = *(const v4u*)(p + 16);
  return f.bf;
}

// B-matrix (32x16 bf16, K x N) per-lane fragment from row-major weights
// W[col][k]: lane (n = lane&15, hi): halves 0-7 -> k = kbase+8*hi+0..7,
// halves 8-15 -> k = kbase+8*hi+16..23 (two global_load_b128).
__device__ __forceinline__ v16bf load_B_frag(const bf16_t* __restrict__ base,
                                             int col, int colStride, int kbase,
                                             int hi) {
  const bf16_t* p = base + (size_t)col * colStride + kbase + 8 * hi;
  Frag f;
  f.q[0] = *(const v4u*)p;
  f.q[1] = *(const v4u*)(p + 16);
  return f.bf;
}

__device__ __forceinline__ v8f wmma_bf16(v16bf a, v16bf b, v8f c) {
  return __builtin_amdgcn_wmma_f32_16x16x32_bf16(false, a, false, b, (short)0,
                                                 c, false, false);
}

__device__ __forceinline__ float sigf(float x) {
  return 1.0f / (1.0f + __expf(-x));
}
__device__ __forceinline__ float tanh_fast(float x) {
  x = fminf(fmaxf(x, -15.0f), 15.0f);
  float e = __expf(2.0f * x);
  return (e - 1.0f) / (e + 1.0f);
}

// ---------------------------------------------------------------------------
// Stage the full 64x1024 bf16 h matrix (128KB) from global into LDS.
// TDM path: one tensor_load_to_lds issued by thread 0 (EXEC is ignored by
// tensor ops); descriptor = 2D tensor, 64 rows x 512 dwords, contiguous.
// Fallback: cooperative 16B-granule copy by all 256 threads.
// The caller must follow with stage_h_wait() (tensorcnt + block barrier).
// ---------------------------------------------------------------------------
__device__ __forceinline__ void stage_h_issue(bf16_t* shh,
                                              const bf16_t* __restrict__ gsrc,
                                              int tid) {
#ifdef USE_TDM
  if (tid == 0) {
    const unsigned lds_off = (unsigned)(size_t)shh; // low 32b = LDS offset
    const unsigned long long ga = (unsigned long long)(size_t)gsrc;
    v4u g0;
    g0.x = 1u;                                    // count=1, user descriptor
    g0.y = lds_off;                               // lds_addr [63:32]
    g0.z = (unsigned)ga;                          // global_addr [95:64]
    g0.w = (unsigned)((ga >> 32) & 0x01FFFFFFull) // global_addr [120:96]
           | (2u << 30);                          // type=2 ("image")
    v8i g1;
    g1[0] = (int)(2u << 16);   // workgroup_mask=0, data_size=2 (4B units)
    g1[1] = (int)(512u << 16); // tensor_dim0 = 512 dwords (bits [79:48])
    g1[2] = (int)(64u << 16);  // tensor_dim0 hi=0; tensor_dim1 = 64
    g1[3] = (int)(512u << 16); // tensor_dim1 hi=0; tile_dim0 = 512
    g1[4] = 64;                // tile_dim1 = 64; tile_dim2 = 0
    g1[5] = 512;               // tensor_dim0_stride = 512 dwords
    g1[6] = 0;
    g1[7] = 0;
    const v4i z4 = {0, 0, 0, 0};                // 2D tensor: groups 2/3 unused
    const v8i z8 = {0, 0, 0, 0, 0, 0, 0, 0};    // 6-arg form extra group
    __builtin_amdgcn_tensor_load_to_lds(g0, g1, z4, z4, z8, 0);
  }
#else
  const v4u* g = (const v4u*)gsrc;
  v4u* s = (v4u*)shh;
#pragma unroll
  for (int i = 0; i < 32; ++i) s[tid + i * NTHREADS] = g[tid + i * NTHREADS];
#endif
}

__device__ __forceinline__ void stage_h_wait(int tid) {
#ifdef USE_TDM
  if (tid == 0) __builtin_amdgcn_s_wait_tensorcnt(0);
#endif
  __syncthreads();
}

// Device-wide sense-reversal barrier (persistent grid of NWG blocks).
__device__ __forceinline__ void grid_barrier(unsigned* __restrict__ cnt,
                                             unsigned* __restrict__ gen) {
  __syncthreads();
  if (threadIdx.x == 0) {
    __threadfence();
    unsigned g = __hip_atomic_load(gen, __ATOMIC_RELAXED,
                                   __HIP_MEMORY_SCOPE_AGENT);
    unsigned a = __hip_atomic_fetch_add(cnt, 1u, __ATOMIC_ACQ_REL,
                                        __HIP_MEMORY_SCOPE_AGENT);
    if (a == (unsigned)(NWG - 1)) {
      __hip_atomic_store(cnt, 0u, __ATOMIC_RELAXED, __HIP_MEMORY_SCOPE_AGENT);
      __hip_atomic_fetch_add(gen, 1u, __ATOMIC_RELEASE,
                             __HIP_MEMORY_SCOPE_AGENT);
    } else {
      while (__hip_atomic_load(gen, __ATOMIC_ACQUIRE,
                               __HIP_MEMORY_SCOPE_AGENT) == g) {
        __builtin_amdgcn_s_sleep(2);
      }
    }
    __threadfence();
  }
  __syncthreads();
}

// ---------------------------------------------------------------------------
// Persistent LSTM kernel. Grid = NWG blocks x 256 threads (8 wave32).
// Each WG owns 32 h-columns; each wave owns (Mtile, 16 cols, all 4 gates).
// Cell states c1/c2 live in registers for all 512 steps. Recurrent h
// matrices are staged into LDS (TDM when available) once per use.
// ---------------------------------------------------------------------------
__global__ __launch_bounds__(NTHREADS, 1) void lstm_persistent(
    const bf16_t* __restrict__ xbf, const bf16_t* __restrict__ wih1,
    const bf16_t* __restrict__ whh1, const bf16_t* __restrict__ wih2,
    const bf16_t* __restrict__ whh2, const bf16_t* __restrict__ wlin,
    const float* __restrict__ bias1, const float* __restrict__ bias2,
    const float* __restrict__ blin, bf16_t* __restrict__ h1buf,
    bf16_t* __restrict__ h2buf, unsigned* __restrict__ bar,
    float* __restrict__ out) {
  extern __shared__ char smem_raw[];
  bf16_t* shh = (bf16_t*)smem_raw;              // 128KB h staging
  float* red = (float*)(smem_raw + SH_H_BYTES); // 8KB: [8][32][8] f32

  const int tid = threadIdx.x;
  const int wg = blockIdx.x; // 0..31
  const int wave = tid >> 5; // 0..7
  const int lane = tid & 31;
  const int n = lane & 15;
  const int hi = lane >> 4;
  const int mtile = wave >> 1;                   // 0..3 (batch tiles of 16)
  const int colbase = wg * 32 + (wave & 1) * 16; // h-column tile base
  const int gcol = colbase + n;                  // this lane's h column
  const int arow = mtile * 16 + n;               // A-fragment row (batch)
  const int mbase = mtile * 16 + 8 * hi;         // C-fragment row base

  float b1g[4], b2g[4];
#pragma unroll
  for (int g = 0; g < 4; ++g) {
    b1g[g] = bias1[g * Hz + gcol];
    b2g[g] = bias2[g * Hz + gcol];
  }

  const int ymt = wg >> 3; // output GEMM: batch tile (0..3)
  const int ynt = wg & 7;  // output GEMM: O tile   (0..7)
  const float yb = blin[ynt * 16 + n];

  v8f c1, c2;
#pragma unroll
  for (int e = 0; e < 8; ++e) {
    c1[e] = 0.0f;
    c2[e] = 0.0f;
  }

#pragma clang loop unroll(disable)
  for (int t = 0; t < Tz; ++t) {
    const int rp = t & 1;
    const bf16_t* h1p = h1buf + (size_t)rp * BH;
    bf16_t* h1n = h1buf + (size_t)(rp ^ 1) * BH;
    const bf16_t* h2p = h2buf + (size_t)rp * BH;
    bf16_t* h2n = h2buf + (size_t)(rp ^ 1) * BH;

    // ---------------- Layer 1: gates = x_t@Wih1^T + h1@Whh1^T + b ----------
    // Kick the h1 stage first; overlap the TDM/copy with the x-part GEMM.
    stage_h_issue(shh, h1p, tid);

    v8f acc[4];
#pragma unroll
    for (int g = 0; g < 4; ++g)
#pragma unroll
      for (int e = 0; e < 8; ++e) acc[g][e] = b1g[g];

#pragma unroll 2
    for (int kb = 0; kb < Iz; kb += 32) {
      v16bf a = load_A_frag(xbf + (size_t)t * Iz, arow, Tz * Iz, kb, hi);
#pragma unroll
      for (int g = 0; g < 4; ++g) {
        v16bf bb = load_B_frag(wih1, g * Hz + gcol, Iz, kb, hi);
        acc[g] = wmma_bf16(a, bb, acc[g]);
      }
    }

    stage_h_wait(tid); // h1p now in LDS

#pragma unroll 2
    for (int kb = 0; kb < Hz; kb += 32) {
      v16bf a = load_A_frag(shh, arow, Hz, kb, hi); // ds_load_b128 x2
#pragma unroll
      for (int g = 0; g < 4; ++g) {
        v16bf bb = load_B_frag(whh1, g * Hz + gcol, Hz, kb, hi);
        acc[g] = wmma_bf16(a, bb, acc[g]);
      }
    }
#pragma unroll
    for (int e = 0; e < 8; ++e) {
      float ig = sigf(acc[0][e]);
      float fg = sigf(acc[1][e]);
      float gg = tanh_fast(acc[2][e]);
      float og = sigf(acc[3][e]);
      float cn = fg * c1[e] + ig * gg;
      c1[e] = cn;
      h1n[(size_t)(mbase + e) * Hz + gcol] = (bf16_t)(og * tanh_fast(cn));
    }

    grid_barrier(bar, bar + 1); // publish h1(t); all shh readers done

    // ---------------- Layer 2: gates = h1@Wih2^T + h2@Whh2^T + b ----------
    stage_h_issue(shh, h1n, tid);

#pragma unroll
    for (int g = 0; g < 4; ++g)
#pragma unroll
      for (int e = 0; e < 8; ++e) acc[g][e] = b2g[g];

    stage_h_wait(tid); // h1(t) in LDS

#pragma unroll 2
    for (int kb = 0; kb < Hz; kb += 32) {
      v16bf a = load_A_frag(shh, arow, Hz, kb, hi);
#pragma unroll
      for (int g = 0; g < 4; ++g) {
        v16bf bb = load_B_frag(wih2, g * Hz + gcol, Hz, kb, hi);
        acc[g] = wmma_bf16(a, bb, acc[g]);
      }
    }

    __syncthreads(); // everyone done reading shh before restaging
    stage_h_issue(shh, h2p, tid);
    stage_h_wait(tid); // h2(t-1) in LDS

#pragma unroll 2
    for (int kb = 0; kb < Hz; kb += 32) {
      v16bf a = load_A_frag(shh, arow, Hz, kb, hi);
#pragma unroll
      for (int g = 0; g < 4; ++g) {
        v16bf bb = load_B_frag(whh2, g * Hz + gcol, Hz, kb, hi);
        acc[g] = wmma_bf16(a, bb, acc[g]);
      }
    }
#pragma unroll
    for (int e = 0; e < 8; ++e) {
      float ig = sigf(acc[0][e]);
      float fg = sigf(acc[1][e]);
      float gg = tanh_fast(acc[2][e]);
      float og = sigf(acc[3][e]);
      float cn = fg * c2[e] + ig * gg;
      c2[e] = cn;
      h2n[(size_t)(mbase + e) * Hz + gcol] = (bf16_t)(og * tanh_fast(cn));
    }

    grid_barrier(bar, bar + 1); // publish h2(t)

    // ---------------- Output: y_t = h2 @ Wlin^T + b ------------------------
    // One 16x16 tile per WG; K=1024 split 4 k-steps per wave, LDS reduce.
    v8f yacc;
#pragma unroll
    for (int e = 0; e < 8; ++e) yacc[e] = 0.0f;
#pragma unroll
    for (int kk = 0; kk < 4; ++kk) {
      const int kb = (wave * 4 + kk) * 32;
      v16bf a = load_A_frag(h2n, ymt * 16 + n, Hz, kb, hi);
      v16bf bb = load_B_frag(wlin, ynt * 16 + n, Hz, kb, hi);
      yacc = wmma_bf16(a, bb, yacc);
    }
#pragma unroll
    for (int e = 0; e < 8; ++e) red[(wave * 32 + lane) * 8 + e] = yacc[e];
    __syncthreads();
    if (tid < 32) {
#pragma unroll
      for (int e = 0; e < 8; ++e) {
        float s = yb;
#pragma unroll
        for (int w = 0; w < 8; ++w) s += red[(w * 32 + lane) * 8 + e];
        const int b = ymt * 16 + 8 * hi + e;
        const int oc = ynt * 16 + n;
        out[((size_t)b * Tz + t) * Oz + oc] = s;
      }
    }
    // red/shh are re-used only after subsequent __syncthreads points.
  }
}

// ---------------------------------------------------------------------------
// Prep kernels
// ---------------------------------------------------------------------------
__global__ void k_cvt_bf16(bf16_t* __restrict__ dst,
                           const float* __restrict__ src, int nelem) {
  int i = blockIdx.x * blockDim.x + threadIdx.x;
  int stride = gridDim.x * blockDim.x;
  for (; i < nelem; i += stride) dst[i] = (bf16_t)src[i];
}

__global__ void k_addf(float* __restrict__ dst, const float* __restrict__ a,
                       const float* __restrict__ b, int nelem) {
  int i = blockIdx.x * blockDim.x + threadIdx.x;
  int stride = gridDim.x * blockDim.x;
  for (; i < nelem; i += stride) dst[i] = a[i] + (b ? b[i] : 0.0f);
}

__global__ void k_zero32(unsigned* __restrict__ p, int nwords) {
  int i = blockIdx.x * blockDim.x + threadIdx.x;
  int stride = gridDim.x * blockDim.x;
  for (; i < nwords; i += stride) p[i] = 0u;
}

// ---------------------------------------------------------------------------
// Host launcher
// ---------------------------------------------------------------------------
extern "C" void kernel_launch(void* const* d_in, const int* in_sizes, int n_in,
                              void* d_out, int out_size, void* d_ws,
                              size_t ws_size, hipStream_t stream) {
  (void)in_sizes; (void)n_in; (void)out_size; (void)ws_size;
  const float* x = (const float*)d_in[0];
  const float* w_ih1 = (const float*)d_in[1];
  const float* w_hh1 = (const float*)d_in[2];
  const float* b_ih1 = (const float*)d_in[3];
  const float* b_hh1 = (const float*)d_in[4];
  const float* w_ih2 = (const float*)d_in[5];
  const float* w_hh2 = (const float*)d_in[6];
  const float* b_ih2 = (const float*)d_in[7];
  const float* b_hh2 = (const float*)d_in[8];
  const float* w_lin = (const float*)d_in[9];
  const float* b_lin = (const float*)d_in[10];
  float* out = (float*)d_out;

  char* ws = (char*)d_ws;
  size_t off = 0;
  auto carve = [&](size_t bytes) -> void* {
    void* p = (void*)(ws + off);
    off += (bytes + 255) & ~(size_t)255;
    return p;
  };

  bf16_t* xbf = (bf16_t*)carve((size_t)Bz * Tz * Iz * 2);
  bf16_t* wih1b = (bf16_t*)carve((size_t)Gz * Iz * 2);
  bf16_t* whh1b = (bf16_t*)carve((size_t)Gz * Hz * 2);
  bf16_t* wih2b = (bf16_t*)carve((size_t)Gz * Hz * 2);
  bf16_t* whh2b = (bf16_t*)carve((size_t)Gz * Hz * 2);
  bf16_t* wlinb = (bf16_t*)carve((size_t)Oz * Hz * 2);
  float* bias1 = (float*)carve((size_t)Gz * 4);
  float* bias2 = (float*)carve((size_t)Gz * 4);
  float* blin = (float*)carve((size_t)Oz * 4);
  // h buffers + barrier carved contiguously so one zero-fill covers them all.
  char* hz_base = (char*)carve((size_t)4 * BH * 2 + 256);
  bf16_t* h1buf = (bf16_t*)hz_base;
  bf16_t* h2buf = (bf16_t*)(hz_base + (size_t)2 * BH * 2);
  unsigned* bar = (unsigned*)(hz_base + (size_t)4 * BH * 2);

  // fp32 -> bf16 conversions (weights stay L2-resident afterwards)
  k_cvt_bf16<<<1024, 256, 0, stream>>>(xbf, x, Bz * Tz * Iz);
  k_cvt_bf16<<<512, 256, 0, stream>>>(wih1b, w_ih1, Gz * Iz);
  k_cvt_bf16<<<1024, 256, 0, stream>>>(whh1b, w_hh1, Gz * Hz);
  k_cvt_bf16<<<1024, 256, 0, stream>>>(wih2b, w_ih2, Gz * Hz);
  k_cvt_bf16<<<1024, 256, 0, stream>>>(whh2b, w_hh2, Gz * Hz);
  k_cvt_bf16<<<128, 256, 0, stream>>>(wlinb, w_lin, Oz * Hz);
  k_addf<<<16, 256, 0, stream>>>(bias1, b_ih1, b_hh1, Gz);
  k_addf<<<16, 256, 0, stream>>>(bias2, b_ih2, b_hh2, Gz);
  k_addf<<<1, 128, 0, stream>>>(blin, b_lin, nullptr, Oz);
  k_zero32<<<256, 256, 0, stream>>>((unsigned*)hz_base,
                                    (int)(((size_t)4 * BH * 2 + 256) / 4));

  // Persistent recurrent kernel: one WG per WGP, device barriers per step,
  // 136KB dynamic LDS (128KB TDM staging + 8KB reduction scratch).
  lstm_persistent<<<NWG, NTHREADS, SMEM_BYTES, stream>>>(
      xbf, wih1b, whh1b, wih2b, whh2b, wlinb, bias1, bias2, blin, h1buf,
      h2buf, bar, out);
}
